// VqVaePlenoxel_35124242546999
// MI455X (gfx1250) — compile-verified, matrix operand order checked by reading, not run on hardware
//
#include <hip/hip_runtime.h>
#include <hip/hip_bf16.h>
#include <math.h>

typedef _Float16 h16;
typedef __attribute__((ext_vector_type(16))) _Float16 v16h;
typedef __attribute__((ext_vector_type(8)))  _Float16 v8h;
typedef __attribute__((ext_vector_type(8)))  float    v8f;

#define RAD_F    1.3f
#define NRAYS    256

// ---------- gfx1250 async global->LDS copy (ASYNCcnt), with sync fallback ----------
#if defined(__has_builtin)
#if __has_builtin(__builtin_amdgcn_global_load_async_to_lds_b128) && \
    __has_builtin(__builtin_amdgcn_s_wait_asynccnt)
#define HAVE_ASYNC_LDS 1
#endif
#endif

typedef int v4i_b __attribute__((vector_size(16)));
typedef __attribute__((address_space(1))) v4i_b* gp128_t;   // global (HIP "__device__") AS1
typedef __attribute__((address_space(3))) v4i_b* lp128_t;   // LDS AS3

__device__ __forceinline__ void async_copy16(const void* gsrc, void* ldst) {
#ifdef HAVE_ASYNC_LDS
  __builtin_amdgcn_global_load_async_to_lds_b128((gp128_t)gsrc, (lp128_t)ldst, 0, 0);
#else
  *(v8h*)ldst = *(const v8h*)gsrc;
#endif
}
__device__ __forceinline__ void async_wait0() {
#ifdef HAVE_ASYNC_LDS
  __builtin_amdgcn_s_wait_asynccnt(0);
#endif
}

// ---- fragment loader: 16x32 f16 tile, row-major source, per CDNA5 WMMA layout ----
// lane l: row = l&15 ; k-base kb + ((l>=16)?8:0) ; elems 0..7 = K kb..kb+7, elems 8..15 = K kb+16..kb+23
__device__ __forceinline__ v16h load_frag(const h16* base, int row, int stride, int kb) {
  const h16* p = base + (size_t)row * stride + kb;
  v8h lo = *(const v8h*)(p);
  v8h hi = *(const v8h*)(p + 16);
  v16h f;
#pragma unroll
  for (int i = 0; i < 8; ++i) { f[i] = lo[i]; f[i + 8] = hi[i]; }
  return f;
}

__device__ __forceinline__ v8f wmma16(v16h a, v16h b, v8f c) {
  return __builtin_amdgcn_wmma_f32_16x16x32_f16(false, a, false, b, (short)0, c, false, false);
}

// ---------------- prep: f16 codebook + norms, f16 transposed MLP weights, zero scalars ----------------
__global__ void k_prep(const float* __restrict__ emb_w, const float* __restrict__ W1,
                       const float* __restrict__ W2, const float* __restrict__ W3,
                       h16* embH, float* norms, h16* wts, float* scal) {
  int tid = threadIdx.x;
  if (blockIdx.x < 2) {
    int r = blockIdx.x * 256 + tid;        // 512 codes
    float s = 0.f;
#pragma unroll 8
    for (int k = 0; k < 64; ++k) { float v = emb_w[r * 64 + k]; s += v * v; embH[r * 64 + k] = (h16)v; }
    norms[r] = s;
  } else {
    h16* w1t = wts;            // [32][128]  W1^T, K padded 112->128
    h16* w2t = wts + 4096;     // [32][32]   W2^T
    h16* w3t = wts + 5120;     // [32][32]   W3^T, N padded 28->32
    for (int i = tid; i < 32 * 128; i += 256) { int n = i >> 7, k = i & 127; w1t[i] = (k < 112) ? (h16)W1[k * 32 + n] : (h16)0.f; }
    for (int i = tid; i < 32 * 32;  i += 256) { int n = i >> 5, k = i & 31;  w2t[i] = (h16)W2[k * 32 + n]; }
    for (int i = tid; i < 32 * 32;  i += 256) { int n = i >> 5, k = i & 31;  w3t[i] = (n < 28) ? (h16)W3[k * 28 + n] : (h16)0.f; }
    for (int i = tid; i < 516; i += 256) scal[i] = 0.f;
  }
}

// ---------------- ray setup: normalized dir, tnear, SH basis ----------------
__global__ void k_rays(const float* __restrict__ rays_o, const float* __restrict__ rays_d,
                       float* __restrict__ rayBuf) {
  int b = threadIdx.x;
  float dx = rays_d[b * 3 + 0], dy = rays_d[b * 3 + 1], dz = rays_d[b * 3 + 2];
  float inv = 1.0f / sqrtf(dx * dx + dy * dy + dz * dz);
  float x = dx * inv, y = dy * inv, z = dz * inv;
  float tn = -3.4e38f;
  float rd3[3] = {x, y, z};
  for (int d = 0; d < 3; ++d) {
    float o = rays_o[b * 3 + d];
    float iv = 1.0f / rd3[d];
    float t1 = (-RAD_F - o) * iv, t2 = (RAD_F - o) * iv;
    tn = fmaxf(tn, fminf(t1, t2));
  }
  tn = fmaxf(tn, 0.f);
  float* R = rayBuf + b * 16;
  R[0] = x; R[1] = y; R[2] = z; R[3] = tn;
  const float C0 = 0.28209479177387814f, C1 = 0.4886025119029199f;
  const float C20 = 1.0925484305920792f, C22 = 0.31539156525252005f, C24 = 0.5462742152960396f;
  R[4] = C0;          R[5] = -C1 * y;     R[6] = C1 * z;
  R[7] = -C1 * x;     R[8] = C20 * x * y; R[9] = -C20 * y * z;
  R[10] = C22 * (2.f * z * z - x * x - y * y);
  R[11] = -C20 * x * z;
  R[12] = C24 * (x * x - y * y);
}

// ---------------- trilinear interp: wave per point, lane handles features e and e+32 ----------------
__global__ void k_interp(const float* __restrict__ rays_o, const float* __restrict__ rayBuf,
                         const float* __restrict__ grid, h16* __restrict__ coarseH,
                         float* __restrict__ rownorm, float* __restrict__ maskf,
                         int NINT, float STEPF) {
  int lane = threadIdx.x & 31;
  int p = blockIdx.x * 8 + (threadIdx.x >> 5);
  int b = p / NINT;
  int j = p - b * NINT;
  const float* R = rayBuf + b * 16;
  float t = R[3] + STEPF * (float)j;
  float pc[3];
#pragma unroll
  for (int d = 0; d < 3; ++d) pc[d] = rays_o[b * 3 + d] + t * R[d];
  float mf = (fabsf(pc[0]) < RAD_F && fabsf(pc[1]) < RAD_F && fabsf(pc[2]) < RAD_F) ? 1.f : 0.f;
  int i0[3], i1[3]; float fr[3];
#pragma unroll
  for (int d = 0; d < 3; ++d) {
    float idx = fminf(fmaxf((pc[d] / RAD_F + 1.f) * 0.5f * 63.f, 0.f), 63.f);
    int ii = (int)floorf(idx);
    i0[d] = ii; fr[d] = idx - (float)ii; i1[d] = min(ii + 1, 63);
  }
  float wx = fr[0], wy = fr[1], wz = fr[2];
  int x0 = i0[0], y0 = i0[1], z0 = i0[2], x1 = i1[0], y1 = i1[1], z1 = i1[2];
  int  off[8] = { z0*4096 + y0*64 + x0, z0*4096 + y0*64 + x1, z0*4096 + y1*64 + x0, z0*4096 + y1*64 + x1,
                  z1*4096 + y0*64 + x0, z1*4096 + y0*64 + x1, z1*4096 + y1*64 + x0, z1*4096 + y1*64 + x1 };
  float wgt[8] = { (1-wx)*(1-wy)*(1-wz), wx*(1-wy)*(1-wz), (1-wx)*wy*(1-wz), wx*wy*(1-wz),
                   (1-wx)*(1-wy)*wz,     wx*(1-wy)*wz,     (1-wx)*wy*wz,     wx*wy*wz };
  const float* g0 = grid + (size_t)lane * 262144;
  const float* g1 = grid + (size_t)(lane + 32) * 262144;
  float v0 = 0.f, v1 = 0.f;
#pragma unroll
  for (int c = 0; c < 8; ++c) { v0 += g0[off[c]] * wgt[c]; v1 += g1[off[c]] * wgt[c]; }
  coarseH[(size_t)p * 64 + lane]      = (h16)v0;
  coarseH[(size_t)p * 64 + lane + 32] = (h16)v1;
  float s = v0 * v0 + v1 * v1;
#pragma unroll
  for (int o = 16; o; o >>= 1) s += __shfl_xor(s, o, 32);
  if (lane == 0) { rownorm[p] = s; maskf[p] = mf; }
}

// ---------------- masked-count reduction ----------------
__global__ void k_redmask(const float* __restrict__ maskf, float* scal, int P) {
  __shared__ float sm[256];
  float s = 0.f;
  for (int i = blockIdx.x * 256 + threadIdx.x; i < P; i += 256 * 256) s += maskf[i];
  sm[threadIdx.x] = s; __syncthreads();
  for (int o = 128; o; o >>= 1) { if (threadIdx.x < (unsigned)o) sm[threadIdx.x] += sm[threadIdx.x + o]; __syncthreads(); }
  if (threadIdx.x == 0) atomicAdd(&scal[1], sm[0]);
}

// ---------------- VQ: WMMA GEMM (codes x points) + streaming argmin ----------------
// Codebook staged to LDS in two 32KB phases via async-to-LDS.
// D[m][n] = dot(emb[m], coarse[n]) - 0.5*||emb[m]||^2  (C-init) ; maximize over m.
__global__ void k_vq(const h16* __restrict__ coarseH, const h16* __restrict__ embH,
                     const float* __restrict__ norms, const float* __restrict__ rownorm,
                     const float* __restrict__ maskf, int* __restrict__ code, float* scal) {
  __shared__ __align__(16) h16   semb[256 * 64];   // 32KB: one phase of the codebook
  __shared__ __align__(16) float snorm[512];       // 2KB: all code norms
  int tid = threadIdx.x;
  int lane = tid & 31, wv = tid >> 5;
  int pt0 = (blockIdx.x * 8 + wv) * 16;
  int r = lane & 15, hi = lane >> 4;
  // B fragments: 16 coarse points, held in registers across all 512 codes
  v16h b0 = load_frag(coarseH, pt0 + r, 64, 0  + hi * 8);
  v16h b1 = load_frag(coarseH, pt0 + r, 64, 32 + hi * 8);
  float best = -3.4e38f; int bestm = 0;
  for (int ph = 0; ph < 2; ++ph) {
    if (ph) __syncthreads();                       // phase-0 reads done before overwrite
    const h16* gsrc = embH + (size_t)ph * 256 * 64;
#pragma unroll
    for (int it = 0; it < 8; ++it) {               // 256 thr x 8 x 16B = 32KB
      int idx = (tid * 8 + it) * 8;
      async_copy16(gsrc + idx, semb + idx);
    }
    if (ph == 0 && tid < 128) async_copy16(norms + tid * 4, snorm + tid * 4);
    async_wait0();
    __syncthreads();
#pragma unroll 2
    for (int mt = 0; mt < 16; ++mt) {
      int m0l = mt * 16;                           // local row in semb
      int mg  = ph * 256 + m0l;                    // global code id
      v16h a0 = load_frag(semb, m0l + r, 64, 0  + hi * 8);
      v16h a1 = load_frag(semb, m0l + r, 64, 32 + hi * 8);
      v8f c;
#pragma unroll
      for (int i = 0; i < 8; ++i) c[i] = -0.5f * snorm[mg + hi * 8 + i];
      c = wmma16(a0, b0, c);
      c = wmma16(a1, b1, c);
#pragma unroll
      for (int i = 0; i < 8; ++i) {
        int m = mg + hi * 8 + i;
        if (c[i] > best) { best = c[i]; bestm = m; }
      }
    }
  }
  // combine the two lane-halves (same point column, code rows split 0-7 / 8-15)
  float ob = __shfl_xor(best, 16, 32);
  int   om = __shfl_xor(bestm, 16, 32);
  if (ob > best || (ob == best && om < bestm)) { best = ob; bestm = om; }
  float e = 0.f;
  if (lane < 16) {
    int p = pt0 + lane;
    code[p] = bestm;
    float mf = maskf[p];
    e = mf * (rownorm[p] - 2.f * best);            // == mf * ||coarse - emb[code]||^2
    atomicAdd(&scal[2 + bestm], mf);
  }
#pragma unroll
  for (int o = 16; o; o >>= 1) e += __shfl_xor(e, o, 32);
  if (lane == 0) atomicAdd(&scal[0], e);
}

// ---------------- MLP (112->32->32->28) fully in WMMA, wave per 16 points ----------------
__global__ void k_mlp(const float* __restrict__ rays_o, const float* __restrict__ rayBuf,
                      const int* __restrict__ code, const float* __restrict__ emb_w,
                      const h16* __restrict__ wts,
                      const float* __restrict__ b1, const float* __restrict__ b2, const float* __restrict__ b3,
                      const float* __restrict__ maskf, float* __restrict__ pt4,
                      int NINT, float STEPF) {
  __shared__ __align__(16) h16 xbuf[8][16 * 128];  // 32KB: per-wave input staging (reused as f32 shd)
  __shared__ __align__(16) h16 hbuf[8][16 * 32];   // 8KB: per-wave hidden activations
  __shared__ __align__(16) h16 swt[6144];          // 12KB: W1^T|W2^T|W3^T staged via async-to-LDS
  int tid = threadIdx.x;
  int lane = tid & 31, wv = tid >> 5;
#pragma unroll
  for (int it = 0; it < 3; ++it) {                 // 256 thr x 3 x 16B = 12KB
    int idx = (tid * 3 + it) * 8;
    async_copy16(wts + idx, swt + idx);
  }
  async_wait0();
  __syncthreads();
  const h16* sw1 = swt;
  const h16* sw2 = swt + 4096;
  const h16* sw3 = swt + 5120;

  int pt0 = (blockIdx.x * 8 + wv) * 16;
  int r = lane & 15, hi = lane >> 4;
  int p = pt0 + r;
  int b = p / NINT;
  int j = p - b * NINT;
  const float* R = rayBuf + b * 16;
  // ---- build input x[128] (quant | pe(fine) | pe(dir) | pad), two lanes per point ----
  if (hi == 0) {
    int cd = code[p];
#pragma unroll 8
    for (int k = 0; k < 64; ++k) xbuf[wv][r * 128 + k] = (h16)emb_w[cd * 64 + k];
  } else {
    float t = R[3] + STEPF * (float)j;
#pragma unroll
    for (int d = 0; d < 3; ++d) {
      float pc  = rays_o[b * 3 + d] + t * R[d];
      float pcc = pc * (64.f / (2.f * RAD_F)) + 32.f;
      float fine = (pcc - floorf(pcc)) * 2.f - 1.f;
      float dir  = R[d];
#pragma unroll
      for (int f = 0; f < 4; ++f) {
        float fq = (float)(1 << f);
        xbuf[wv][r * 128 + 64 + d * 8 + f]     = (h16)sinf(fine * fq);
        xbuf[wv][r * 128 + 64 + d * 8 + 4 + f] = (h16)cosf(fine * fq);
        xbuf[wv][r * 128 + 88 + d * 8 + f]     = (h16)sinf(dir * fq);
        xbuf[wv][r * 128 + 88 + d * 8 + 4 + f] = (h16)cosf(dir * fq);
      }
    }
#pragma unroll
    for (int k = 112; k < 128; ++k) xbuf[wv][r * 128 + k] = (h16)0.f;
  }
  // ---- layer 1: x(16x128) @ W1(128x32) ----
  v16h a0 = load_frag(&xbuf[wv][0], r, 128, 0  + hi * 8);
  v16h a1 = load_frag(&xbuf[wv][0], r, 128, 32 + hi * 8);
  v16h a2 = load_frag(&xbuf[wv][0], r, 128, 64 + hi * 8);
  v16h a3 = load_frag(&xbuf[wv][0], r, 128, 96 + hi * 8);
#pragma unroll
  for (int nt = 0; nt < 2; ++nt) {
    int n = nt * 16 + r;
    v8f c;
#pragma unroll
    for (int i = 0; i < 8; ++i) c[i] = b1[n];
    c = wmma16(a0, load_frag(sw1, n, 128, 0  + hi * 8), c);
    c = wmma16(a1, load_frag(sw1, n, 128, 32 + hi * 8), c);
    c = wmma16(a2, load_frag(sw1, n, 128, 64 + hi * 8), c);
    c = wmma16(a3, load_frag(sw1, n, 128, 96 + hi * 8), c);
#pragma unroll
    for (int i = 0; i < 8; ++i) { int m = i + hi * 8; hbuf[wv][m * 32 + n] = (h16)fmaxf(c[i], 0.f); }
  }
  // ---- layer 2: h1(16x32) @ W2(32x32) ----
  v16h h1 = load_frag(&hbuf[wv][0], r, 32, hi * 8);
  v8f c2[2];
#pragma unroll
  for (int nt = 0; nt < 2; ++nt) {
    int n = nt * 16 + r;
    v8f c;
#pragma unroll
    for (int i = 0; i < 8; ++i) c[i] = b2[n];
    c2[nt] = wmma16(h1, load_frag(sw2, n, 32, hi * 8), c);
  }
#pragma unroll
  for (int nt = 0; nt < 2; ++nt) {
    int n = nt * 16 + r;
#pragma unroll
    for (int i = 0; i < 8; ++i) { int m = i + hi * 8; hbuf[wv][m * 32 + n] = (h16)fmaxf(c2[nt][i], 0.f); }
  }
  // ---- layer 3: h2(16x32) @ W3(32x28 pad 32); shd overlays xbuf (dead after layer-1 loads) ----
  float* sb = (float*)&xbuf[wv][0];                // 16x32 f32 = 2KB per wave
  v16h h2 = load_frag(&hbuf[wv][0], r, 32, hi * 8);
#pragma unroll
  for (int nt = 0; nt < 2; ++nt) {
    int n = nt * 16 + r;
    v8f c;
#pragma unroll
    for (int i = 0; i < 8; ++i) c[i] = (n < 28) ? b3[n] : 0.f;
    c = wmma16(h2, load_frag(sw3, n, 32, hi * 8), c);
#pragma unroll
    for (int i = 0; i < 8; ++i) { int m = i + hi * 8; sb[m * 32 + n] = c[i]; }
  }
  // ---- per-point epilogue: sigma + SH->rgb ----
  if (hi == 0) {
    float mf = maskf[p];
    float sg = fmaxf(sb[r * 32 + 27], 0.f) * mf;
    float rgb[3];
#pragma unroll
    for (int c3 = 0; c3 < 3; ++c3) {
      float s = 0.f;
#pragma unroll
      for (int k = 0; k < 9; ++k) s += R[4 + k] * sb[r * 32 + c3 * 9 + k];
      rgb[c3] = fminf(fmaxf(s * mf + 0.5f, 0.f), 1.f);
    }
    ((float4*)pt4)[p] = make_float4(rgb[0], rgb[1], rgb[2], sg);
  }
}

// ---------------- per-ray volume integration ----------------
__global__ void k_integrate(const float* __restrict__ rayBuf, const float* __restrict__ pt4,
                            float* __restrict__ out, int NINT, float STEPF) {
  int b = threadIdx.x;
  float tn = rayBuf[b * 16 + 3];
  float T = 1.f, rr = 0.f, gg = 0.f, bb = 0.f, dep = 0.f, wsum = 0.f;
  const float4* pv = (const float4*)pt4;
  for (int i = 0; i < NINT; ++i) {
    float4 v = pv[(size_t)b * NINT + i];
    float alpha = 1.f - expf(-v.w * STEPF);
    float w = alpha * T;
    rr += w * v.x; gg += w * v.y; bb += w * v.z;
    dep += w * (tn + STEPF * (float)i);
    wsum += w;
    T *= (1.f - alpha + 1e-10f);
  }
  float bg = 1.f - wsum;
  out[b * 3 + 0] = rr + bg;
  out[b * 3 + 1] = gg + bg;
  out[b * 3 + 2] = bb + bg;
  out[768 + b] = dep;
}

// ---------------- scalars: commit loss + perplexity ----------------
__global__ void k_final(const float* __restrict__ scal, float* __restrict__ out) {
  float cnt = fmaxf(scal[1], 1.f);
  out[1024] = 0.25f * scal[0] / (cnt * 64.f);
  float H = 0.f;
  for (int i = 0; i < 512; ++i) { float avg = scal[2 + i] / cnt; H += avg * logf(avg + 1e-10f); }
  out[1025] = expf(-H);
}

extern "C" void kernel_launch(void* const* d_in, const int* in_sizes, int n_in,
                              void* d_out, int out_size, void* d_ws, size_t ws_size,
                              hipStream_t stream) {
  const float* rays_o = (const float*)d_in[0];
  const float* rays_d = (const float*)d_in[1];
  const float* grid   = (const float*)d_in[3];
  const float* emb_w  = (const float*)d_in[4];
  const float* W1 = (const float*)d_in[5];
  const float* b1 = (const float*)d_in[6];
  const float* W2 = (const float*)d_in[7];
  const float* b2 = (const float*)d_in[8];
  const float* W3 = (const float*)d_in[9];
  const float* b3 = (const float*)d_in[10];
  float* out = (float*)d_out;

  const double STEP_D = 2.0 * 1.3 / 63.0 / 8.0;
  const int NINT = (int)(sqrt(3.0) * 2.0 * 1.3 / STEP_D) - 1;   // 871
  const int P = NRAYS * NINT;                                   // 222976 (mult of 128)
  const float STEPF = (float)STEP_D;

  char* w = (char*)d_ws;
  float* rayBuf  = (float*)w; w += 256 * 16 * 4;
  float* norms   = (float*)w; w += 512 * 4;
  float* scal    = (float*)w; w += 516 * 4;           // [0]=e_sum [1]=cnt [2..513]=hist
  h16*   embH    = (h16*)w;   w += 512 * 64 * 2;
  h16*   wts     = (h16*)w;   w += 6144 * 2;          // W1^T | W2^T | W3^T contiguous
  h16*   coarseH = (h16*)w;   w += (size_t)P * 64 * 2;
  float* rownorm = (float*)w; w += (size_t)P * 4;
  float* maskf   = (float*)w; w += (size_t)P * 4;
  int*   code    = (int*)w;   w += (size_t)P * 4;
  float* pt4     = (float*)w; w += (size_t)P * 16;

  k_prep<<<3, 256, 0, stream>>>(emb_w, W1, W2, W3, embH, norms, wts, scal);
  k_rays<<<1, 256, 0, stream>>>(rays_o, rays_d, rayBuf);
  k_interp<<<P / 8, 256, 0, stream>>>(rays_o, rayBuf, grid, coarseH, rownorm, maskf, NINT, STEPF);
  k_redmask<<<256, 256, 0, stream>>>(maskf, scal, P);
  k_vq<<<P / 128, 256, 0, stream>>>(coarseH, embH, norms, rownorm, maskf, code, scal);
  k_mlp<<<P / 128, 256, 0, stream>>>(rays_o, rayBuf, code, emb_w, wts,
                                     b1, b2, b3, maskf, pt4, NINT, STEPF);
  k_integrate<<<1, 256, 0, stream>>>(rayBuf, pt4, out, NINT, STEPF);
  k_final<<<1, 1, 0, stream>>>(scal, out);
}
